// VectorQuantizerEMA_15960098472212
// MI455X (gfx1250) — compile-verified, matrix operand order checked by reading, not run on hardware
//
#include <hip/hip_runtime.h>
#include <hip/hip_bf16.h>

// ---------------------------------------------------------------------------
// VQ-VAE vector quantizer, CDNA5 (gfx1250) wave32 + v_wmma_f32_16x16x32_bf16
//
// inputs:  d_in[0] = inputs   [32, 64, 64, 64]  f32 (BCHW)
//          d_in[1] = embedding[512, 64]         f32
// outputs: d_out (f32, concatenated):
//          loss[131072] | q_st BCHW[8388608] | perplexity[1] | idx[131072]
// ---------------------------------------------------------------------------

typedef __attribute__((ext_vector_type(16))) __bf16 v16bf;
typedef __attribute__((ext_vector_type(8)))  __bf16 v8bf;
typedef __attribute__((ext_vector_type(8)))  float  v8f;

// exact pointee type expected by the async-copy builtin (per hipcc diagnostic):
// GCC-style vector: __attribute__((__vector_size__(16))) int
typedef int gv4i __attribute__((__vector_size__(16)));

#define AS1 __attribute__((address_space(1)))
#define AS3 __attribute__((address_space(3)))

#if defined(__has_builtin)
# if __has_builtin(__builtin_amdgcn_global_load_async_to_lds_b128)
#  define HAVE_ASYNC_COPY 1
# endif
#endif
#ifndef HAVE_ASYNC_COPY
# define HAVE_ASYNC_COPY 0
#endif

#define NUM_CODES   512
#define CDIM        64
#define HW          4096            // 64*64
#define NTOK        131072          // 32*4096
#define TOK_PER_BLK 256
#define NBLK        512             // NTOK / TOK_PER_BLK

// dynamic LDS layout (bytes); all 16B aligned
#define EBF_OFF   0                      // bf16 [512][72]  = 73728
#define XBF_OFF   73728                  // bf16 [256][72]  = 36864
#define EN_OFF    110592                 // f32  [512]      =  2048
#define BIDX_OFF  112640                 // i32  [256]      =  1024
#define SMEM_BYTES 113664

// ---------------------------------------------------------------------------
// Kernel 1: embedding f32 -> bf16 + row norms; zero histogram (every launch)
// ---------------------------------------------------------------------------
__global__ void vq_prep(const float* __restrict__ emb,
                        __bf16* __restrict__ ebf,
                        float*  __restrict__ enorm,
                        unsigned int* __restrict__ counts) {
  int k = blockIdx.x * blockDim.x + threadIdx.x;
  if (k >= NUM_CODES) return;
  const float* row = emb + k * CDIM;
  float s = 0.f;
#pragma unroll
  for (int c = 0; c < CDIM; ++c) {
    float v = row[c];
    s += v * v;
    ebf[k * CDIM + c] = (__bf16)v;
  }
  enorm[k]  = s;
  counts[k] = 0u;
}

// ---------------------------------------------------------------------------
// Kernel 2: main — argmin distances via bf16 WMMA, gather, loss, histogram
// ---------------------------------------------------------------------------
__global__ __launch_bounds__(256)
void vq_main(const float* __restrict__ in,        // BCHW f32
             const float* __restrict__ emb,       // [512][64] f32
             const __bf16* __restrict__ ebf,      // [512][64] bf16
             const float* __restrict__ enorm,     // [512] f32
             unsigned int* __restrict__ counts,
             float* __restrict__ out_loss,
             float* __restrict__ out_q,
             float* __restrict__ out_idx) {
  extern __shared__ char smem[];
  __bf16* sE   = (__bf16*)(smem + EBF_OFF);   // [512][72] padded (conflict-free b128)
  __bf16* sX   = (__bf16*)(smem + XBF_OFF);   // [256][72]
  float*  sEn  = (float*) (smem + EN_OFF);    // [512]
  int*    sIdx = (int*)   (smem + BIDX_OFF);  // [256]

  const int tid = threadIdx.x;
  const int b   = blockIdx.x >> 4;              // 16 blocks per batch image
  const int hw0 = (blockIdx.x & 15) * TOK_PER_BLK;
  const float* xbase = in + (size_t)b * (CDIM * HW) + hw0;

  // keep the fp32 codebook warm for the gather pass (global_prefetch_b8)
  __builtin_prefetch(emb + tid * CDIM, 0, 0);
  __builtin_prefetch(emb + (tid + 256) * CDIM, 0, 0);

  // ---- stage bf16 codebook into LDS, row stride 72 bf16 (= 144 B)
#if HAVE_ASYNC_COPY
  {
    unsigned int ldsE = (unsigned int)(uintptr_t)(void*)sE;
    for (int it = 0; it < 16; ++it) {
      int item = tid + 256 * it;                // 512 rows * 8 chunks of 16B
      int row  = item >> 3, part = item & 7;
      __builtin_amdgcn_global_load_async_to_lds_b128(
          (AS1 gv4i*)(uintptr_t)(const void*)(ebf + row * 64 + part * 8),
          (AS3 gv4i*)(unsigned int)(ldsE + row * 144 + part * 16),
          0, 0);
    }
# if __has_builtin(__builtin_amdgcn_s_wait_asynccnt)
    __builtin_amdgcn_s_wait_asynccnt(0);
# else
    asm volatile("s_wait_asynccnt 0" ::: "memory");
# endif
  }
#else
  {
    const uint4* g = (const uint4*)ebf;         // 512 rows * 8 chunks = 4096
    for (int it = 0; it < 16; ++it) {
      int item = tid + 256 * it;
      int row  = item >> 3, part = item & 7;
      uint4 v = g[item];
      *(uint4*)(sE + row * 72 + part * 8) = v;
    }
  }
#endif
#pragma unroll
  for (int it = 0; it < 2; ++it) sEn[tid + 256 * it] = enorm[tid + 256 * it];

  // ---- stage X tile (bf16): per iteration one channel row, 256 contiguous f32
  for (int c = 0; c < CDIM; ++c) {
    float v = xbase[c * HW + tid];
    sX[tid * 72 + c] = (__bf16)v;
  }
  __syncthreads();

  // ---- per-wave: two 16-token tiles, WMMA argmin over 32 code tiles
  {
    const int wave = tid >> 5;
    const int lane = tid & 31;
    const int half = lane >> 4;                 // ISA A-layout: lane half picks K offset
    const int lm   = lane & 15;
    const int kb   = half ? 8 : 0;

    // A fragments, ISA 7.12.2 16-bit 16x32 layout:
    // VGPR0-3: K = kb..kb+7, VGPR4-7: K = kb+16..kb+23  (two ds_load_b128 each)
    const __bf16* xpa = sX + (wave * 32 + lm) * 72;        // tile a
    const __bf16* xpb = xpa + 16 * 72;                     // tile b
    v8bf t0, t1;
    t0 = *(const v8bf*)(xpa + kb);      t1 = *(const v8bf*)(xpa + kb + 16);
    v16bf A0a = __builtin_shufflevector(t0, t1, 0,1,2,3,4,5,6,7,8,9,10,11,12,13,14,15);
    t0 = *(const v8bf*)(xpa + kb + 32); t1 = *(const v8bf*)(xpa + kb + 48);
    v16bf A1a = __builtin_shufflevector(t0, t1, 0,1,2,3,4,5,6,7,8,9,10,11,12,13,14,15);
    t0 = *(const v8bf*)(xpb + kb);      t1 = *(const v8bf*)(xpb + kb + 16);
    v16bf A0b = __builtin_shufflevector(t0, t1, 0,1,2,3,4,5,6,7,8,9,10,11,12,13,14,15);
    t0 = *(const v8bf*)(xpb + kb + 32); t1 = *(const v8bf*)(xpb + kb + 48);
    v16bf A1b = __builtin_shufflevector(t0, t1, 0,1,2,3,4,5,6,7,8,9,10,11,12,13,14,15);

    float bestVa[8], bestVb[8];
    int   bestIa[8], bestIb[8];
#pragma unroll
    for (int r = 0; r < 8; ++r) {
      bestVa[r] = 3.4e38f; bestIa[r] = 0;
      bestVb[r] = 3.4e38f; bestIb[r] = 0;
    }

    for (int kt = 0; kt < 32; ++kt) {
      int code = kt * 16 + lm;                  // this lane's column (B mirrored layout)
      const __bf16* ep = sE + code * 72;
      v8bf b0lo = *(const v8bf*)(ep + kb);
      v8bf b0hi = *(const v8bf*)(ep + kb + 16);
      v8bf b1lo = *(const v8bf*)(ep + kb + 32);
      v8bf b1hi = *(const v8bf*)(ep + kb + 48);
      v16bf B0 = __builtin_shufflevector(b0lo, b0hi, 0,1,2,3,4,5,6,7,8,9,10,11,12,13,14,15);
      v16bf B1 = __builtin_shufflevector(b1lo, b1hi, 0,1,2,3,4,5,6,7,8,9,10,11,12,13,14,15);

      v8f z = {0.f,0.f,0.f,0.f,0.f,0.f,0.f,0.f};
      // two independent accumulation chains; B0/B1 each feed two WMMAs
      v8f aa = __builtin_amdgcn_wmma_f32_16x16x32_bf16(false, A0a, false, B0,
                                                       (short)0, z,  false, false);
      v8f ab = __builtin_amdgcn_wmma_f32_16x16x32_bf16(false, A0b, false, B0,
                                                       (short)0, z,  false, false);
      aa = __builtin_amdgcn_wmma_f32_16x16x32_bf16(false, A1a, false, B1,
                                                   (short)0, aa, false, false);
      ab = __builtin_amdgcn_wmma_f32_16x16x32_bf16(false, A1b, false, B1,
                                                   (short)0, ab, false, false);
      float en = sEn[code];
#pragma unroll
      for (int r = 0; r < 8; ++r) {
        // argmin unaffected by ||x||^2 -> score = ||e||^2 - 2 x.e
        float sa = __builtin_fmaf(-2.f, aa[r], en);
        float sb = __builtin_fmaf(-2.f, ab[r], en);
        if (sa < bestVa[r]) { bestVa[r] = sa; bestIa[r] = code; }
        if (sb < bestVb[r]) { bestVb[r] = sb; bestIb[r] = code; }
      }
    }

    // width-16 (val,idx) min-reduce across lanes sharing a token row
#pragma unroll
    for (int r = 0; r < 8; ++r) {
      float va = bestVa[r]; int ia = bestIa[r];
      float vb = bestVb[r]; int ib = bestIb[r];
#pragma unroll
      for (int m = 8; m >= 1; m >>= 1) {
        float v2 = __shfl_xor(va, m, 16);
        int   i2 = __shfl_xor(ia, m, 16);
        if (v2 < va || (v2 == va && i2 < ia)) { va = v2; ia = i2; }  // first-index ties
        float v3 = __shfl_xor(vb, m, 16);
        int   i3 = __shfl_xor(ib, m, 16);
        if (v3 < vb || (v3 == vb && i3 < ib)) { vb = v3; ib = i3; }
      }
      if (lm == 0) {
        sIdx[wave * 32 + half * 8 + r]      = ia;
        sIdx[wave * 32 + 16 + half * 8 + r] = ib;
      }
    }
  }
  __syncthreads();

  // ---- outputs: thread `tid` owns token `tid` -> loss in a register,
  //      quantized gather with coalesced BCHW stores
  {
    const int k = sIdx[tid];
    const float* erow = emb + k * CDIM;
    float* qbase = out_q + (size_t)b * (CDIM * HW) + hw0;
    float acc = 0.f;
#pragma unroll 8
    for (int c = 0; c < CDIM; ++c) {
      float q = erow[c];
      qbase[c * HW + tid] = q;                  // q_st value == quantized
      float d = q - xbase[c * HW + tid];        // x re-read (L2-resident)
      acc = __builtin_fmaf(d, d, acc);
    }
    const int n = b * HW + hw0 + tid;
    out_loss[n] = 0.25f * (acc * (1.0f / 64.0f));
    out_idx[n]  = (float)k;
    atomicAdd(&counts[k], 1u);                  // usage histogram
  }
}

// ---------------------------------------------------------------------------
// Kernel 3: perplexity = exp(-sum p log(p + 1e-10))
// ---------------------------------------------------------------------------
__global__ void vq_perp(const unsigned int* __restrict__ counts,
                        float* __restrict__ out_perp) {
  __shared__ float red[NUM_CODES];
  int k = threadIdx.x;
  float p = (float)counts[k] * (1.0f / (float)NTOK);
  red[k] = -p * logf(p + 1e-10f);
  __syncthreads();
  for (int s = NUM_CODES / 2; s > 0; s >>= 1) {
    if (k < s) red[k] += red[k + s];
    __syncthreads();
  }
  if (k == 0) out_perp[0] = expf(red[0]);
}

// ---------------------------------------------------------------------------
extern "C" void kernel_launch(void* const* d_in, const int* in_sizes, int n_in,
                              void* d_out, int out_size, void* d_ws, size_t ws_size,
                              hipStream_t stream) {
  const float* in  = (const float*)d_in[0];
  const float* emb = (const float*)d_in[1];
  float* out = (float*)d_out;

  // workspace: bf16 codebook | f32 norms | u32 histogram
  __bf16*       ebf    = (__bf16*)d_ws;                               // 65536 B
  float*        enorm  = (float*)((char*)d_ws + 65536);               //  2048 B
  unsigned int* counts = (unsigned int*)((char*)d_ws + 65536 + 2048); //  2048 B

  float* out_loss = out;
  float* out_q    = out + NTOK;
  float* out_perp = out + NTOK + (size_t)NTOK * CDIM;
  float* out_idx  = out_perp + 1;

  // allow >48KB dynamic LDS (CDNA5: up to 320KB/WG); host-side config, capture-safe
  (void)hipFuncSetAttribute((const void*)vq_main,
                            hipFuncAttributeMaxDynamicSharedMemorySize, SMEM_BYTES);

  vq_prep<<<2, 256, 0, stream>>>(emb, ebf, enorm, counts);
  vq_main<<<NBLK, 256, SMEM_BYTES, stream>>>(in, emb, ebf, enorm, counts,
                                             out_loss, out_q, out_idx);
  vq_perp<<<1, NUM_CODES, 0, stream>>>(counts, out_perp);
}
